// DualTimeConstantAdaptation_21844203667493
// MI455X (gfx1250) — matrix operand order repaired and checked by reading, not compiled, not run on hardware
//
#include <hip/hip_runtime.h>

typedef __attribute__((ext_vector_type(16))) _Float16 v16h;
typedef __attribute__((ext_vector_type(8)))  _Float16 v8h;
typedef __attribute__((ext_vector_type(8)))  float    v8f;
typedef __attribute__((ext_vector_type(4)))  int      v4i;

#define T_IN      48000
#define KTAPS     4800
#define TOUT      (T_IN + KTAPS - 1)      // 52799
#define ROWS      512
#define PAD_FRONT (KTAPS - 1)             // 4799
#define PADT      57856                   // covers max padded index 57807
#define NCHUNK    (KTAPS / 16)            // 300
#define NTILES    3300                    // ceil(TOUT/16)
#define WAVES_PER_WG   4
#define NT_PER_WAVE    4
#define TILES_PER_WG   (WAVES_PER_WG * NT_PER_WAVE)                 // 16
#define TBLOCKS        ((NTILES + TILES_PER_WG - 1) / TILES_PER_WG) // 207
#define XS_BYTES  ((size_t)ROWS * PADT * 2)                         // ~59.2 MB in d_ws
#define BB_BYTES  ((size_t)NCHUNK * 32 * 16 * 2)                    // 300 KB band matrices

// LDS window: 16 rows x 288 halves of signal per chunk, stride padded to 296
#define WIN_HALVES 288
#define LDS_STRIDE 296
#define GRANULES   (16 * WIN_HALVES / 8)  // 576 16-byte granules per window

#if __has_builtin(__builtin_amdgcn_global_load_async_to_lds_b128)
#define HAVE_ASYNC_LDS 1
#endif

__device__ __forceinline__ void copy16_to_lds(const _Float16* g, _Float16* l) {
#ifdef HAVE_ASYNC_LDS
    __builtin_amdgcn_global_load_async_to_lds_b128(
        (__attribute__((address_space(1))) v4i*)g,
        (__attribute__((address_space(3))) v4i*)l, 0, 0);
#else
    *(v8h*)l = *(const v8h*)g;
#endif
}

__device__ __forceinline__ void wait_async_copies() {
#ifdef HAVE_ASYNC_LDS
#if __has_builtin(__builtin_amdgcn_s_wait_asynccnt)
    __builtin_amdgcn_s_wait_asynccnt(0);
#else
    asm volatile("s_wait_asynccnt 0x0" ::: "memory");
#endif
#endif
}

// ---- prep 1: padded softplus(x) -> f16 signal in workspace -----------------
__global__ void prep_xs_kernel(const float* __restrict__ x, _Float16* __restrict__ xs) {
    unsigned i = blockIdx.x * blockDim.x + threadIdx.x;
    if (i >= (unsigned)ROWS * PADT) return;
    unsigned r = i / PADT;
    int      t = (int)(i - r * PADT) - PAD_FRONT;
    float v = 0.0f;
    if (t >= 0 && t < T_IN) {
        float xv = x[(size_t)r * T_IN + t];
        v = fmaxf(xv, 0.0f) + log1pf(__expf(-fabsf(xv)));   // stable softplus
    }
    xs[i] = (_Float16)v;
}

// ---- prep 2: banded-Toeplitz kernel matrices, one 32x16 f16 per chunk ------
// WMMA B layout: lane l holds column n = l%16, halves b[s] = W[j,n] = w[k0 + j - n], j = table-order k (8h+s / 16+8h+s-8).
__global__ void prep_band_kernel(_Float16* __restrict__ bband) {
    unsigned i = blockIdx.x * blockDim.x + threadIdx.x;
    if (i >= (unsigned)NCHUNK * 32) return;
    int c    = i >> 5;
    int lane = i & 31;
    int k0   = c * 16;
    double Sr = (1.0 - exp(-(double)KTAPS / 32.0))  / (1.0 - exp(-1.0 / 32.0));
    double Ss = (1.0 - exp(-(double)KTAPS / 960.0)) / (1.0 - exp(-1.0 / 960.0));
    float invS = (float)(1.0 / (Sr + Ss));
    v16h out;
#pragma unroll
    for (int s = 0; s < 16; ++s) {
        int n = lane & 15;
        int j = (s < 8) ? ((lane >> 4) * 8 + s) : (16 + (lane >> 4) * 8 + (s - 8));
        int d = j - n;
        float w = 0.0f;
        if (d >= 0 && d < 16) {
            float k = (float)(k0 + d);
            w = (__expf(-k * (1.0f / 32.0f)) + __expf(-k * (1.0f / 960.0f))) * invS;
        }
        out[s] = (_Float16)w;
    }
    ((v16h*)bband)[i] = out;
}

// ---- main: async-LDS staged implicit-GEMM via v_wmma_f32_16x16x32_f16 ------
__global__ __launch_bounds__(WAVES_PER_WG * 32)
void conv_wmma_kernel(const _Float16* __restrict__ xs,
                      const _Float16* __restrict__ bband,
                      float* __restrict__ out) {
    __shared__ alignas(16) _Float16 smem[2][16 * LDS_STRIDE];

    const int rowTile = blockIdx.x;                 // 0..31
    const int tBlock  = blockIdx.y;                 // 0..206
    const int tid     = threadIdx.x;
    const int wave    = tid >> 5;
    const int lane    = tid & 31;
    const int hiHalf  = lane >> 4;
    const int halfSel = hiHalf << 3;                // A-layout K offset (0 or 8)
    const int rowIn   = lane & 15;

    const int t0_wg = tBlock * TILES_PER_WG * 16;   // block's padded time base
    const int ntLoc = wave * NT_PER_WAVE;           // this wave's first tile in block

    const size_t xsRowStride = (size_t)PADT;

    v8f acc[NT_PER_WAVE] = {};

    // prologue: stage chunk 0 window into buffer 0
    {
        const int winStart = t0_wg;                 // k0 = 0
#pragma unroll
        for (int it = 0; it < 5; ++it) {
            int g = tid + it * (WAVES_PER_WG * 32);
            if (g < GRANULES) {
                int row   = g / (WIN_HALVES / 8);
                int col16 = g - row * (WIN_HALVES / 8);
                const _Float16* src = xs + (size_t)(rowTile * 16 + row) * xsRowStride
                                         + winStart + col16 * 8;
                _Float16* dst = &smem[0][row * LDS_STRIDE + col16 * 8];
                copy16_to_lds(src, dst);
            }
        }
    }

    for (int c = 0; c < NCHUNK; ++c) {
        wait_async_copies();      // my granules of chunk c landed
        __syncthreads();          // everyone's landed; prior buffer free for reuse

        // stage chunk c+1 into the other buffer (overlaps with compute below)
        if (c + 1 < NCHUNK) {
            const int winStart = t0_wg + (c + 1) * 16;
            _Float16* buf = smem[(c + 1) & 1];
#pragma unroll
            for (int it = 0; it < 5; ++it) {
                int g = tid + it * (WAVES_PER_WG * 32);
                if (g < GRANULES) {
                    int row   = g / (WIN_HALVES / 8);
                    int col16 = g - row * (WIN_HALVES / 8);
                    const _Float16* src = xs + (size_t)(rowTile * 16 + row) * xsRowStride
                                             + winStart + col16 * 8;
                    copy16_to_lds(src, &buf[row * LDS_STRIDE + col16 * 8]);
                }
            }
        }

        // B operand: 32x16 Toeplitz band for this chunk (small, L2-resident)
        const v16h bm = ((const v16h*)bband)[c * 32 + lane];
        const _Float16* buf = smem[c & 1];

#pragma unroll
        for (int nt = 0; nt < NT_PER_WAVE; ++nt) {
            const int off = (ntLoc + nt) * 16;      // local window offset
            union { v16h v; v8h h[2]; } a;
            a.h[0] = *(const v8h*)&buf[rowIn * LDS_STRIDE + off + halfSel];
            a.h[1] = *(const v8h*)&buf[rowIn * LDS_STRIDE + off + 16 + halfSel];
            acc[nt] = __builtin_amdgcn_wmma_f32_16x16x32_f16(
                false, a.v, false, bm, (short)0, acc[nt], false, false);
        }
        __syncthreads();          // all reads of buf[c&1] done before it is refilled
    }

    // D layout: lane(0-15)=n ; VGPR r -> row r (lanes<16) / r+8 (lanes>=16)
    const size_t row0 = (size_t)(rowTile * 16 + (hiHalf << 3));
    const int tileBase = tBlock * TILES_PER_WG + ntLoc;
#pragma unroll
    for (int nt = 0; nt < NT_PER_WAVE; ++nt) {
        const int tg = (tileBase + nt) * 16 + rowIn;
        if (tg < TOUT) {
#pragma unroll
            for (int r = 0; r < 8; ++r) {
                out[(row0 + r) * (size_t)TOUT + tg] = acc[nt][r];
            }
        }
    }
}

extern "C" void kernel_launch(void* const* d_in, const int* in_sizes, int n_in,
                              void* d_out, int out_size, void* d_ws, size_t ws_size,
                              hipStream_t stream) {
    (void)in_sizes; (void)n_in; (void)out_size; (void)ws_size;
    const float* x   = (const float*)d_in[0];
    float*       out = (float*)d_out;
    _Float16*    xs    = (_Float16*)d_ws;                       // padded f16 signal
    _Float16*    bband = (_Float16*)((char*)d_ws + XS_BYTES);   // band matrices

    {
        unsigned total = (unsigned)ROWS * PADT;
        prep_xs_kernel<<<(total + 255) / 256, 256, 0, stream>>>(x, xs);
    }
    {
        unsigned total = (unsigned)NCHUNK * 32;
        prep_band_kernel<<<(total + 255) / 256, 256, 0, stream>>>(bband);
    }
    {
        dim3 grid(ROWS / 16, TBLOCKS);                          // 32 x 207 workgroups
        conv_wmma_kernel<<<grid, WAVES_PER_WG * 32, 0, stream>>>(xs, bband, out);
    }
}
